// DecoderLayer_79989470921247
// MI455X (gfx1250) — matrix operand order repaired and checked by reading, not compiled, not run on hardware
//
#include <hip/hip_runtime.h>
#include <hip/hip_fp16.h>
#include <math.h>
#include <stdint.h>

typedef __attribute__((ext_vector_type(16))) _Float16 v16h;
typedef __attribute__((ext_vector_type(8)))  float    v8f;
typedef __attribute__((ext_vector_type(4)))  unsigned int u32x4;
typedef __attribute__((ext_vector_type(8)))  int      i32x8;
typedef __attribute__((ext_vector_type(4)))  int      i32x4;

#define WMMA_F16(a, b, c) \
  __builtin_amdgcn_wmma_f32_16x16x32_f16(false, (a), false, (b), (short)0, (c), false, false)

#if __has_builtin(__builtin_amdgcn_tensor_load_to_lds) && __has_builtin(__builtin_amdgcn_s_wait_tensorcnt)
#define USE_TDM 1
#else
#define USE_TDM 0
#endif

static constexpr int   NN = 4, SS = 100, CC = 4, TT = 32, DD = 256, HH = 8, DKK = 32, FF = 1024;
static constexpr int   MTOK = NN * SS * CC * TT;           // 51200 tokens
static constexpr float INV_SCALE = 0.17677669529663687f;   // 1/sqrt(32)

#if USE_TDM
// ---------------------------------------------------------------------------
// Issue one TDM 2-D tile load: global (rows x 32 f32, row stride strideElems)
// -> LDS at ldsByteAddr, DMA-padded to row stride 34 f32 (conflict-free).
// D# layout per CDNA5 ISA ch.8 (group0 128b, group1 256b; groups 2/3 zero).
// ---------------------------------------------------------------------------
__device__ __forceinline__ void tdm_load_tile(unsigned ldsByteAddr, const float* gp,
                                              int tensorK, int strideElems, int rows)
{
  const unsigned long long ga = (unsigned long long)(uintptr_t)gp;
  u32x4 g0;
  g0[0] = 1u;                                               // count=1 (valid user D#)
  g0[1] = ldsByteAddr;                                      // lds_addr
  g0[2] = (unsigned)(ga & 0xffffffffu);                     // global_addr[31:0]
  g0[3] = (unsigned)((ga >> 32) & 0x01ffffffu) | (2u << 30);// global_addr[56:32] | type=2
  i32x8 g1;
  // data_size=2 (4B) | pad_enable | pad_interval=4 (32 DW) | pad_amount=1 (2 DW)
  g1[0] = (2 << 16) | (1 << 20) | (4 << 22) | (1 << 25);
  g1[1] = (int)(((unsigned)tensorK & 0xffffu) << 16);       // tensor_dim0[15:0]
  g1[2] = (int)((((unsigned)tensorK >> 16) & 0xffffu)       // tensor_dim0[31:16]
        | (((unsigned)rows & 0xffffu) << 16));              // tensor_dim1[15:0]
  g1[3] = (int)(32u << 16);                                 // tensor_dim1[31:16]=0 | tile_dim0=32
  g1[4] = (int)((unsigned)rows & 0xffffu);                  // tile_dim1 = rows (tile_dim2=0)
  g1[5] = strideElems;                                      // tensor_dim0_stride[31:0]
  g1[6] = 0;
  g1[7] = 0;
  const i32x4 z4 = {0, 0, 0, 0};
#if defined(__clang_major__) && __clang_major__ >= 23
  const i32x8 z8 = {};
  __builtin_amdgcn_tensor_load_to_lds(g0, g1, z4, z4, z8, 0);
#else
  __builtin_amdgcn_tensor_load_to_lds(g0, g1, z4, z4, 0);
#endif
}
#endif

// ---------------------------------------------------------------------------
// LDS-tiled WMMA GEMM, double-buffered: Out = A[M,K] @ W[Nout,K]^T + bias.
// Block = 256 threads (8 waves) -> 128(M) x 64(N) macro-tile; wave owns 32x32.
// K-chunks of 32 f32 staged in LDS (TDM DMA when available, manual otherwise),
// row stride 34 floats (17*2 banks -> conflict-free fragment gathers).
// ---------------------------------------------------------------------------
template <int HAS_RES, int RELU>
__global__ __launch_bounds__(256) void gemm_tiled_kernel(
    const float* __restrict__ A, const float* __restrict__ W,
    const float* __restrict__ bias, const float* __restrict__ res,
    float* __restrict__ Out, int M, int Nout, int K)
{
  __shared__ float sA[2][128][34];   // 34816 B
  __shared__ float sB[2][64][34];    // 17408 B

  const int tid  = threadIdx.x;
  const int lane = tid & 31;
  const int wv   = tid >> 5;
  const int wm   = wv & 3;           // 32-row band
  const int wn   = wv >> 2;          // 32-col band
  const int r    = lane & 15;
  const int hi   = lane >> 4;

  const int tilesN = Nout >> 6;
  const int tm = blockIdx.x / tilesN;
  const int tn = blockIdx.x % tilesN;
  const int m0 = tm << 7;
  const int n0 = tn << 6;

  v8f acc[2][2] = {{{}, {}}, {{}, {}}};

  auto compute = [&](int p) {
    v16h a0, a1, b0, b1;
#pragma unroll
    for (int i = 0; i < 16; i += 2) {
      const int ka = (i & 7) + ((i >> 3) << 4) + hi * 8;   // A-layout K index (pairs)
      const float2 fa0 = *(const float2*)&sA[p][wm * 32 + r][ka];
      const float2 fa1 = *(const float2*)&sA[p][wm * 32 + 16 + r][ka];
      a0[i] = (_Float16)fa0.x; a0[i + 1] = (_Float16)fa0.y;
      a1[i] = (_Float16)fa1.x; a1[i + 1] = (_Float16)fa1.y;
      const int kb = (i & 7) + ((i >> 3) << 4) + hi * 8;   // B-layout K index (pairs)
      const float2 fb0 = *(const float2*)&sB[p][wn * 32 + r][kb];
      const float2 fb1 = *(const float2*)&sB[p][wn * 32 + 16 + r][kb];
      b0[i] = (_Float16)fb0.x; b0[i + 1] = (_Float16)fb0.y;
      b1[i] = (_Float16)fb1.x; b1[i + 1] = (_Float16)fb1.y;
    }
    acc[0][0] = WMMA_F16(a0, b0, acc[0][0]);
    acc[0][1] = WMMA_F16(a0, b1, acc[0][1]);
    acc[1][0] = WMMA_F16(a1, b0, acc[1][0]);
    acc[1][1] = WMMA_F16(a1, b1, acc[1][1]);
  };

#if USE_TDM
  auto issue = [&](int buf, int kk) {
    if (wv == 0) {   // wave-uniform: one wave drives the DMA engine
      tdm_load_tile((unsigned)(uintptr_t)&sA[buf][0][0], A + (size_t)m0 * K + kk, K, K, 128);
      tdm_load_tile((unsigned)(uintptr_t)&sB[buf][0][0], W + (size_t)n0 * K + kk, K, K, 64);
    }
  };
  issue(0, 0);
  if (wv == 0) __builtin_amdgcn_s_wait_tensorcnt(0);
  __syncthreads();
  for (int kk = 0; kk < K; kk += 32) {
    const int p = (kk >> 5) & 1;
    const bool next = (kk + 32) < K;
    if (next) issue(1 - p, kk + 32);      // DMA next chunk while we compute
    compute(p);
    if (next && wv == 0) __builtin_amdgcn_s_wait_tensorcnt(0);
    __syncthreads();
  }
#else
  // manual double-buffered staging
  const int la_row = tid >> 1;            // 0..127
  const int la_k   = (tid & 1) * 16;      // 0 / 16
  const int lb_n   = tid >> 2;            // 0..63
  const int lb_k   = (tid & 3) * 8;       // 0/8/16/24
  float4 ra[4], rb[2];
  auto gload = [&](int kk) {
    const float4* ap = (const float4*)(A + (size_t)(m0 + la_row) * K + kk + la_k);
#pragma unroll
    for (int q = 0; q < 4; ++q) ra[q] = ap[q];
    const float4* bp = (const float4*)(W + (size_t)(n0 + lb_n) * K + kk + lb_k);
#pragma unroll
    for (int q = 0; q < 2; ++q) rb[q] = bp[q];
    __builtin_prefetch(ap + 8, 0, 1);
  };
  auto commit = [&](int buf) {
#pragma unroll
    for (int q = 0; q < 4; ++q) {
      *(float2*)&sA[buf][la_row][la_k + q * 4]     = make_float2(ra[q].x, ra[q].y);
      *(float2*)&sA[buf][la_row][la_k + q * 4 + 2] = make_float2(ra[q].z, ra[q].w);
    }
#pragma unroll
    for (int q = 0; q < 2; ++q) {
      *(float2*)&sB[buf][lb_n][lb_k + q * 4]     = make_float2(rb[q].x, rb[q].y);
      *(float2*)&sB[buf][lb_n][lb_k + q * 4 + 2] = make_float2(rb[q].z, rb[q].w);
    }
  };
  gload(0); commit(0);
  __syncthreads();
  for (int kk = 0; kk < K; kk += 32) {
    const int p = (kk >> 5) & 1;
    const bool next = (kk + 32) < K;
    if (next) gload(kk + 32);             // loads in flight during compute
    compute(p);
    if (next) commit(1 - p);
    __syncthreads();
  }
#endif

  // ---- epilogue (straight-line; templated) ----
#pragma unroll
  for (int sn = 0; sn < 2; ++sn) {
    const int col = n0 + wn * 32 + sn * 16 + r;
    const float bv = bias ? bias[col] : 0.0f;
#pragma unroll
    for (int sm = 0; sm < 2; ++sm) {
#pragma unroll
      for (int j = 0; j < 8; ++j) {
        const size_t row = (size_t)m0 + wm * 32 + sm * 16 + j + hi * 8;
        const size_t idx = row * Nout + col;
        float v = acc[sm][sn][j] + bv;
        if (HAS_RES) v += res[idx];
        if (RELU)    v = v > 0.0f ? v : 0.0f;
        Out[idx] = v;
      }
    }
  }
}

// ---------------------------------------------------------------------------
// Time attention: one wave per (n,s,c,h). T=32, dk=32. 8 WMMAs per wave.
// ---------------------------------------------------------------------------
__global__ __launch_bounds__(128) void attn_time_kernel(
    const float* __restrict__ Q, const float* __restrict__ Kb,
    const float* __restrict__ V, const unsigned char* __restrict__ mask,
    float* __restrict__ O)
{
  __shared__ float    sS[4][32][33];
  __shared__ _Float16 sP[4][32][32];

  const int lane = threadIdx.x & 31;
  const int w    = threadIdx.x >> 5;
  const int gid  = blockIdx.x * 4 + w;      // over N*S*C*H = 12800 (exact)
  const int h    = gid & 7;
  const int rem  = gid >> 3;                // (n*S+s)*C+c
  const size_t t0   = (size_t)rem * TT;
  const int    hcol = h * DKK;
  const int r  = lane & 15;
  const int hi = lane >> 4;

  // ---- S = Q K^T * inv_scale ----
  {
    v16h a0, a1;
    const float* q0 = Q + (t0 + r) * DD + hcol;
    const float* q1 = Q + (t0 + 16 + r) * DD + hcol;
#pragma unroll
    for (int i = 0; i < 16; ++i) {
      const int k = (i & 7) + ((i >> 3) << 4) + hi * 8;
      a0[i] = (_Float16)q0[k];
      a1[i] = (_Float16)q1[k];
    }
#pragma unroll
    for (int tu = 0; tu < 2; ++tu) {
      v16h b;
      const float* kp = Kb + (t0 + tu * 16 + r) * DD + hcol + hi * 8;
#pragma unroll
      for (int i = 0; i < 16; ++i) b[i] = (_Float16)kp[(i & 7) + ((i >> 3) << 4)];
      v8f acc0 = {}, acc1 = {};
      acc0 = WMMA_F16(a0, b, acc0);
      acc1 = WMMA_F16(a1, b, acc1);
#pragma unroll
      for (int j = 0; j < 8; ++j) {
        sS[w][j + hi * 8][tu * 16 + r]      = acc0[j] * INV_SCALE;
        sS[w][16 + j + hi * 8][tu * 16 + r] = acc1[j] * INV_SCALE;
      }
    }
  }
  __syncthreads();

  // ---- softmax: lane owns row `lane` ----
  {
    float sv[32];
    float mx = -1e30f;
#pragma unroll
    for (int u = 0; u < 32; ++u) {
      float s = sS[w][lane][u];
      if (mask && mask[lane * TT + u]) s = -1e30f;
      sv[u] = s; mx = s > mx ? s : mx;
    }
    float sum = 0.0f;
#pragma unroll
    for (int u = 0; u < 32; ++u) { sv[u] = __expf(sv[u] - mx); sum += sv[u]; }
    const float rs = 1.0f / sum;
#pragma unroll
    for (int u = 0; u < 32; ++u) sP[w][lane][u] = (_Float16)(sv[u] * rs);
  }
  __syncthreads();

  // ---- O = P V ----
  {
    v16h a0, a1;
#pragma unroll
    for (int i = 0; i < 16; ++i) {
      const int k = (i & 7) + ((i >> 3) << 4) + hi * 8;   // key index u
      a0[i] = sP[w][r][k];
      a1[i] = sP[w][16 + r][k];
    }
#pragma unroll
    for (int td = 0; td < 2; ++td) {
      v16h b;
#pragma unroll
      for (int i = 0; i < 16; ++i)
        b[i] = (_Float16)V[(t0 + (i & 7) + ((i >> 3) << 4) + hi * 8) * DD + hcol + td * 16 + r];
      v8f acc0 = {}, acc1 = {};
      acc0 = WMMA_F16(a0, b, acc0);
      acc1 = WMMA_F16(a1, b, acc1);
#pragma unroll
      for (int j = 0; j < 8; ++j) {
        O[(t0 + j + hi * 8) * DD + hcol + td * 16 + r]      = acc0[j];
        O[(t0 + 16 + j + hi * 8) * DD + hcol + td * 16 + r] = acc1[j];
      }
    }
  }
}

// ---------------------------------------------------------------------------
// Channel attention: attend over C=4. One thread per (n,h,s,t).
// ---------------------------------------------------------------------------
__global__ __launch_bounds__(256) void attn_chan_kernel(
    const float* __restrict__ Q, const float* __restrict__ Kb,
    const float* __restrict__ V, float* __restrict__ O)
{
  int idx = blockIdx.x * 256 + threadIdx.x;      // N*H*S*T = 102400
  if (idx >= NN * HH * SS * TT) return;
  const int t = idx & 31; int tmp = idx >> 5;
  const int s = tmp % SS; tmp /= SS;
  const int h = tmp & 7;  const int n = tmp >> 3;
  const int hcol = h * DKK;
  size_t tok[CC];
#pragma unroll
  for (int c = 0; c < CC; ++c)
    tok[c] = ((size_t)((n * SS + s) * CC + c) * TT + t) * DD + hcol;

  float sc[CC][CC];
#pragma unroll
  for (int ci = 0; ci < CC; ++ci)
#pragma unroll
    for (int cj = 0; cj < CC; ++cj) {
      float d = 0.0f;
      for (int k = 0; k < DKK; ++k) d += Q[tok[ci] + k] * Kb[tok[cj] + k];
      sc[ci][cj] = d * INV_SCALE;
    }
#pragma unroll
  for (int ci = 0; ci < CC; ++ci) {
    float mx = sc[ci][0];
#pragma unroll
    for (int cj = 1; cj < CC; ++cj) mx = sc[ci][cj] > mx ? sc[ci][cj] : mx;
    float sum = 0.0f;
#pragma unroll
    for (int cj = 0; cj < CC; ++cj) { sc[ci][cj] = __expf(sc[ci][cj] - mx); sum += sc[ci][cj]; }
    const float rs = 1.0f / sum;
    for (int k = 0; k < DKK; ++k) {
      float o = 0.0f;
#pragma unroll
      for (int cj = 0; cj < CC; ++cj) o += sc[ci][cj] * rs * V[tok[cj] + k];
      O[tok[ci] + k] = o;
    }
  }
}

// ---------------------------------------------------------------------------
// Space attention: attend over S=100. One block per (n,h,c,t); K/V/S in LDS.
// ---------------------------------------------------------------------------
__global__ __launch_bounds__(128) void attn_space_kernel(
    const float* __restrict__ Q, const float* __restrict__ Kb,
    const float* __restrict__ V, float* __restrict__ O)
{
  __shared__ float sK[SS * DKK];
  __shared__ float sV[SS * DKK];
  __shared__ float sS[SS * SS];

  int b = blockIdx.x;                // N*H*C*T = 4096
  const int t = b & 31; b >>= 5;
  const int c = b & 3;  b >>= 2;
  const int h = b & 7;  const int n = b >> 3;
  const int hcol = h * DKK;

  auto tokOf = [&](int s) { return ((size_t)((n * SS + s) * CC + c) * TT + t) * DD; };

  for (int i = threadIdx.x; i < SS * DKK; i += 128) {
    const int s_ = i >> 5, d = i & 31;
    sK[i] = Kb[tokOf(s_) + hcol + d];
    sV[i] = V [tokOf(s_) + hcol + d];
  }
  __syncthreads();

  const int srow = threadIdx.x;
  if (srow < SS) {
    float q[DKK];
#pragma unroll
    for (int d = 0; d < DKK; ++d) q[d] = Q[tokOf(srow) + hcol + d];
    float mx = -1e30f;
    for (int u = 0; u < SS; ++u) {
      float dot = 0.0f;
#pragma unroll
      for (int d = 0; d < DKK; ++d) dot += q[d] * sK[u * DKK + d];
      dot *= INV_SCALE;
      sS[srow * SS + u] = dot;
      mx = dot > mx ? dot : mx;
    }
    float sum = 0.0f;
    for (int u = 0; u < SS; ++u) {
      float e = __expf(sS[srow * SS + u] - mx);
      sS[srow * SS + u] = e; sum += e;
    }
    const float rs = 1.0f / sum;
    float o[DKK];
#pragma unroll
    for (int d = 0; d < DKK; ++d) o[d] = 0.0f;
    for (int u = 0; u < SS; ++u) {
      const float p = sS[srow * SS + u] * rs;
#pragma unroll
      for (int d = 0; d < DKK; ++d) o[d] += p * sV[u * DKK + d];
    }
#pragma unroll
    for (int d = 0; d < DKK; ++d) O[tokOf(srow) + hcol + d] = o[d];
  }
}

// ---------------------------------------------------------------------------
// LayerNorm with fused residuals: y = LN(x + s1*r1 + s2*r2)*g + b
// ---------------------------------------------------------------------------
__global__ __launch_bounds__(256) void ln_kernel(
    const float* __restrict__ x, const float* __restrict__ r1,
    const float* __restrict__ r2, const float* __restrict__ s1p,
    const float* __restrict__ s2p, const float* __restrict__ g,
    const float* __restrict__ bta, float* __restrict__ out)
{
  const int lane = threadIdx.x & 31;
  const int wv   = threadIdx.x >> 5;
  const size_t tok = (size_t)blockIdx.x * 8 + wv;
  if (tok >= (size_t)MTOK) return;
  const size_t base = tok * DD;
  const float s1 = s1p ? s1p[0] : 1.0f;
  const float s2 = s2p ? s2p[0] : 1.0f;

  float v[8];
  float sum = 0.0f;
#pragma unroll
  for (int i = 0; i < 8; ++i) {
    const int d = lane + i * 32;
    float t = x[base + d];
    if (r1) t += s1 * r1[base + d];
    if (r2) t += s2 * r2[base + d];
    v[i] = t; sum += t;
  }
#pragma unroll
  for (int o = 16; o > 0; o >>= 1) sum += __shfl_xor(sum, o, 32);
  const float mu = sum * (1.0f / DD);
  float var = 0.0f;
#pragma unroll
  for (int i = 0; i < 8; ++i) { const float d = v[i] - mu; var += d * d; }
#pragma unroll
  for (int o = 16; o > 0; o >>= 1) var += __shfl_xor(var, o, 32);
  const float inv = rsqrtf(var * (1.0f / DD) + 1e-5f);
#pragma unroll
  for (int i = 0; i < 8; ++i) {
    const int d = lane + i * 32;
    out[base + d] = (v[i] - mu) * inv * g[d] + bta[d];
  }
}

// ---------------------------------------------------------------------------
// PhyCell kernels (frames (N,C,100,256); x_t gathered from token-major X).
// ---------------------------------------------------------------------------
__global__ __launch_bounds__(256) void zero_kernel(float* p, long n) {
  long i = (long)blockIdx.x * 256 + threadIdx.x;
  if (i < n) p[i] = 0.0f;
}

__global__ __launch_bounds__(256) void phy_gate_kernel(
    const float* __restrict__ X, int t, const float* __restrict__ H,
    const float* __restrict__ wg, const float* __restrict__ bg,
    float* __restrict__ Kg)
{
  int idx = blockIdx.x * 256 + threadIdx.x;      // 409600
  if (idx >= NN * CC * SS * DD) return;
  const int x = idx & 255; int r = idx >> 8;
  const int y = r % SS; r /= SS;
  const int co = r & 3; const int n = r >> 2;
  float sum = bg[co];
  for (int ci = 0; ci < 8; ++ci)
    for (int ky = -1; ky <= 1; ++ky) {
      const int yy = y + ky; if (yy < 0 || yy >= SS) continue;
      for (int kx = -1; kx <= 1; ++kx) {
        const int xx = x + kx; if (xx < 0 || xx >= DD) continue;
        float v;
        if (ci < 4) v = X[((size_t)((n * SS + yy) * CC + ci) * TT + t) * DD + xx];
        else        v = H[((size_t)(n * CC + (ci - 4)) * SS + yy) * DD + xx];
        sum += v * wg[((co * 8 + ci) * 3 + (ky + 1)) * 3 + (kx + 1)];
      }
    }
  Kg[idx] = 1.0f / (1.0f + __expf(-sum));
}

__global__ __launch_bounds__(256) void phy_convf1_kernel(
    const float* __restrict__ H, const float* __restrict__ wf1,
    const float* __restrict__ bf1, float* __restrict__ F1)
{
  int idx = blockIdx.x * 256 + threadIdx.x;      // 5017600
  if (idx >= NN * 49 * SS * DD) return;
  const int x = idx & 255; int r = idx >> 8;
  const int y = r % SS; r /= SS;
  const int co = r % 49; const int n = r / 49;
  float sum = bf1[co];
  for (int ci = 0; ci < CC; ++ci)
    for (int ky = -3; ky <= 3; ++ky) {
      const int yy = y + ky; if (yy < 0 || yy >= SS) continue;
      for (int kx = -3; kx <= 3; ++kx) {
        const int xx = x + kx; if (xx < 0 || xx >= DD) continue;
        sum += H[((size_t)(n * CC + ci) * SS + yy) * DD + xx]
             * wf1[((co * CC + ci) * 7 + (ky + 3)) * 7 + (kx + 3)];
      }
    }
  F1[idx] = sum;
}

__global__ __launch_bounds__(256) void phy_gnstats_kernel(
    const float* __restrict__ F1, float* __restrict__ stats)
{
  __shared__ float s1[256], s2[256];
  const int n = blockIdx.x / 7, g = blockIdx.x % 7;    // 28 blocks
  const int c0 = g * 7;
  const int per = 7 * SS * DD;                          // 179200
  float a = 0.0f, b = 0.0f;
  for (int i = threadIdx.x; i < per; i += 256) {
    const int c = c0 + i / (SS * DD);
    const int rem = i % (SS * DD);
    const float v = F1[(size_t)(n * 49 + c) * SS * DD + rem];
    a += v; b += v * v;
  }
  s1[threadIdx.x] = a; s2[threadIdx.x] = b;
  __syncthreads();
  for (int o = 128; o > 0; o >>= 1) {
    if (threadIdx.x < o) { s1[threadIdx.x] += s1[threadIdx.x + o]; s2[threadIdx.x] += s2[threadIdx.x + o]; }
    __syncthreads();
  }
  if (threadIdx.x == 0) {
    const float mean = s1[0] / per;
    const float var  = s2[0] / per - mean * mean;
    stats[(n * 7 + g) * 2]     = mean;
    stats[(n * 7 + g) * 2 + 1] = rsqrtf(var + 1e-5f);
  }
}

__global__ __launch_bounds__(256) void phy_f2_kernel(
    const float* __restrict__ F1, const float* __restrict__ stats,
    const float* __restrict__ gng, const float* __restrict__ gnb,
    const float* __restrict__ wf2, const float* __restrict__ bf2,
    float* __restrict__ Fb)
{
  int idx = blockIdx.x * 256 + threadIdx.x;      // 409600
  if (idx >= NN * CC * SS * DD) return;
  const int x = idx & 255; int r = idx >> 8;
  const int y = r % SS; r /= SS;
  const int co = r & 3; const int n = r >> 2;
  float sum = bf2[co];
  for (int c = 0; c < 49; ++c) {
    const int g = c / 7;
    const float m  = stats[(n * 7 + g) * 2];
    const float rs = stats[(n * 7 + g) * 2 + 1];
    const float v = (F1[((size_t)(n * 49 + c) * SS + y) * DD + x] - m) * rs * gng[c] + gnb[c];
    sum += v * wf2[co * 49 + c];
  }
  Fb[idx] = sum;
}

__global__ __launch_bounds__(256) void phy_grad_kernel(
    const float* __restrict__ H, const float* __restrict__ gxw,
    const float* __restrict__ gyw, float* __restrict__ G)
{
  int idx = blockIdx.x * 256 + threadIdx.x;      // 409600
  if (idx >= NN * CC * SS * DD) return;
  const int x = idx & 255; int r = idx >> 8;
  const int y = r % SS; r /= SS;
  const int c = r & 3; const int n = r >> 2;
  float gx = 0.0f, gy = 0.0f;
  for (int ci = 0; ci < CC; ++ci) {
    for (int k = -1; k <= 1; ++k) {
      const int xx = x + k;
      if (xx >= 0 && xx < DD)
        gx += H[((size_t)(n * CC + ci) * SS + y) * DD + xx] * gxw[(c * CC + ci) * 3 + (k + 1)];
      const int yy = y + k;
      if (yy >= 0 && yy < SS)
        gy += H[((size_t)(n * CC + ci) * SS + yy) * DD + x] * gyw[(c * CC + ci) * 3 + (k + 1)];
    }
  }
  G[idx] = sqrtf(gx * gx + gy * gy + 1e-12f);
}

__global__ __launch_bounds__(256) void phy_bnstats_kernel(
    const float* __restrict__ G, float* __restrict__ stats)
{
  __shared__ float s1[256], s2[256];
  const int c = blockIdx.x;                      // 4 blocks
  const int per = NN * SS * DD;                  // 102400
  float a = 0.0f, b = 0.0f;
  for (int i = threadIdx.x; i < per; i += 256) {
    const int n = i / (SS * DD);
    const int rem = i % (SS * DD);
    const float v = G[(size_t)(n * CC + c) * SS * DD + rem];
    a += v; b += v * v;
  }
  s1[threadIdx.x] = a; s2[threadIdx.x] = b;
  __syncthreads();
  for (int o = 128; o > 0; o >>= 1) {
    if (threadIdx.x < o) { s1[threadIdx.x] += s1[threadIdx.x + o]; s2[threadIdx.x] += s2[threadIdx.x + o]; }
    __syncthreads();
  }
  if (threadIdx.x == 0) {
    const float mean = s1[0] / per;
    const float var  = s2[0] / per - mean * mean;
    stats[56 + c * 2]     = mean;
    stats[56 + c * 2 + 1] = rsqrtf(var + 1e-5f);
  }
}

__global__ __launch_bounds__(256) void phy_combine_kernel(
    const float* __restrict__ X, int t, float* __restrict__ H,
    const float* __restrict__ Fb, const float* __restrict__ G,
    const float* __restrict__ Kg, const float* __restrict__ stats,
    const float* __restrict__ bng, const float* __restrict__ bnb,
    float* __restrict__ Pout)
{
  int idx = blockIdx.x * 256 + threadIdx.x;      // 409600
  if (idx >= NN * CC * SS * DD) return;
  const int x = idx & 255; int r = idx >> 8;
  const int y = r % SS; r /= SS;
  const int c = r & 3; const int n = r >> 2;
  const float om = (G[idx] - stats[56 + c * 2]) * stats[56 + c * 2 + 1] * bng[c] + bnb[c];
  const float ht = H[idx] + Fb[idx] + om;
  const size_t tokd = ((size_t)((n * SS + y) * CC + c) * TT + t) * DD + x;
  const float xt = X[tokd];
  const float nh = ht + Kg[idx] * (xt - ht);
  H[idx]     = nh;
  Pout[tokd] = nh;
}

// ---------------------------------------------------------------------------
// Host orchestration
// ---------------------------------------------------------------------------
extern "C" void kernel_launch(void* const* d_in, const int* in_sizes, int n_in,
                              void* d_out, int out_size, void* d_ws, size_t ws_size,
                              hipStream_t stream)
{
  (void)in_sizes; (void)n_in; (void)out_size; (void)ws_size;
  const float* X0  = (const float*)d_in[0];
  const float* MEM = (const float*)d_in[1];
  const unsigned char* MASK = (const unsigned char*)d_in[2];
  auto F = [&](int i) { return (const float*)d_in[i]; };

  const int PB_TIME = 3, PB_CHAN = 11, PB_SPACE = 19, PB_ENC = 27;
  const int PB_TIME2 = 35, PB_CHAN2 = 43, PB_SPACE2 = 51;
  const float *w1 = F(59), *b1 = F(60), *w2 = F(61), *b2 = F(62);
  const float *p1 = F(63), *p2 = F(64);
  const float *wg = F(65), *bg = F(66), *wf1 = F(67), *bf1 = F(68);
  const float *gng = F(69), *gnb = F(70), *wf2 = F(71), *bf2 = F(72);
  const float *gxw = F(73), *gyw = F(74), *bng = F(75), *bnb = F(76);
  const float *ln0g = F(77), *ln0b = F(78), *ln1g = F(79), *ln1b = F(80);
  const float *ln2g = F(81), *ln2b = F(82), *ln3g = F(83), *ln3b = F(84);

  const size_t MD = (size_t)MTOK * DD;
  float* ws  = (float*)d_ws;
  float* W0  = ws;            // current x
  float* Wq  = W0 + MD;
  float* Wk  = Wq + MD;
  float* Wv  = Wk + MD;
  float* Wat = Wv + MD;
  float* Wm  = Wat + MD;
  float* Wn  = Wm + MD;
  float* Wp  = Wn + MD;       // phy output (token-major)
  float* Wff = Wp + MD;       // M x 1024
  float* HBUF  = Wff + (size_t)MTOK * FF;
  float* F1BUF = HBUF + (size_t)NN * CC * SS * DD;
  float* KGBUF = F1BUF + (size_t)NN * 49 * SS * DD;
  float* GBUF  = KGBUF + (size_t)NN * CC * SS * DD;
  float* FBBUF = GBUF  + (size_t)NN * CC * SS * DD;
  float* STATS = FBBUF + (size_t)NN * CC * SS * DD;   // 64 floats

  auto gemm = [&](const float* A, const float* W, const float* bias,
                  const float* res, float* Out, int Nout, int K, int relu) {
    const int blocks = (MTOK / 128) * (Nout / 64);
    if (res)
      gemm_tiled_kernel<1, 0><<<blocks, 256, 0, stream>>>(A, W, bias, res, Out, MTOK, Nout, K);
    else if (relu)
      gemm_tiled_kernel<0, 1><<<blocks, 256, 0, stream>>>(A, W, bias, res, Out, MTOK, Nout, K);
    else
      gemm_tiled_kernel<0, 0><<<blocks, 256, 0, stream>>>(A, W, bias, res, Out, MTOK, Nout, K);
  };
  auto ln = [&](const float* x, const float* r1, const float* r2,
                const float* s1, const float* s2,
                const float* g, const float* b, float* out) {
    ln_kernel<<<MTOK / 8, 256, 0, stream>>>(x, r1, r2, s1, s2, g, b, out);
  };
  auto mha_time = [&](const float* qin, const float* kvin, int pb,
                      const unsigned char* msk, const float* res, float* out) {
    gemm(qin,  F(pb + 0), F(pb + 1), nullptr, Wq, DD, DD, 0);
    gemm(kvin, F(pb + 2), F(pb + 3), nullptr, Wk, DD, DD, 0);
    gemm(kvin, F(pb + 4), F(pb + 5), nullptr, Wv, DD, DD, 0);
    attn_time_kernel<<<(NN * SS * CC * HH) / 4, 128, 0, stream>>>(Wq, Wk, Wv, msk, Wat);
    gemm(Wat, F(pb + 6), F(pb + 7), res, out, DD, DD, 0);
  };
  auto mha_chan = [&](const float* in, int pb, const float* res, float* out) {
    gemm(in, F(pb + 0), F(pb + 1), nullptr, Wq, DD, DD, 0);
    gemm(in, F(pb + 2), F(pb + 3), nullptr, Wk, DD, DD, 0);
    gemm(in, F(pb + 4), F(pb + 5), nullptr, Wv, DD, DD, 0);
    attn_chan_kernel<<<(NN * HH * SS * TT + 255) / 256, 256, 0, stream>>>(Wq, Wk, Wv, Wat);
    gemm(Wat, F(pb + 6), F(pb + 7), res, out, DD, DD, 0);
  };
  auto mha_space = [&](const float* in, int pb, const float* res, float* out) {
    gemm(in, F(pb + 0), F(pb + 1), nullptr, Wq, DD, DD, 0);
    gemm(in, F(pb + 2), F(pb + 3), nullptr, Wk, DD, DD, 0);
    gemm(in, F(pb + 4), F(pb + 5), nullptr, Wv, DD, DD, 0);
    attn_space_kernel<<<NN * HH * CC * TT, 128, 0, stream>>>(Wq, Wk, Wv, Wat);
    gemm(Wat, F(pb + 6), F(pb + 7), res, out, DD, DD, 0);
  };

  // ---- Stage A: x = LN0(x + dst(x)) ----
  mha_time(X0, X0, PB_TIME, MASK, nullptr, Wm);
  mha_chan(Wm, PB_CHAN, nullptr, Wn);
  mha_space(Wn, PB_SPACE, nullptr, Wm);
  ln(X0, Wm, nullptr, nullptr, nullptr, ln0g, ln0b, W0);

  // ---- Stage B: x = LN1(x + enc_attn(x, memory)) ----
  mha_time(W0, MEM, PB_ENC, nullptr, nullptr, Wm);
  ln(W0, Wm, nullptr, nullptr, nullptr, ln1g, ln1b, W0);

  // ---- Stage C: dst2(x) -> Wm ; phy(x) -> Wp ----
  mha_time(W0, W0, PB_TIME2, MASK, nullptr, Wm);
  mha_chan(Wm, PB_CHAN2, Wm, Wn);        // + m residual (fused)
  mha_space(Wn, PB_SPACE2, Wn, Wm);      // + n residual (fused)

  {
    const long hsz = (long)NN * CC * SS * DD;
    zero_kernel<<<(int)((hsz + 255) / 256), 256, 0, stream>>>(HBUF, hsz);
    const int eb = (int)((hsz + 255) / 256);
    for (int t = 0; t < TT; ++t) {
      phy_gate_kernel  <<<eb, 256, 0, stream>>>(W0, t, HBUF, wg, bg, KGBUF);
      phy_convf1_kernel<<<(NN * 49 * SS * DD + 255) / 256, 256, 0, stream>>>(HBUF, wf1, bf1, F1BUF);
      phy_gnstats_kernel<<<28, 256, 0, stream>>>(F1BUF, STATS);
      phy_f2_kernel    <<<eb, 256, 0, stream>>>(F1BUF, STATS, gng, gnb, wf2, bf2, FBBUF);
      phy_grad_kernel  <<<eb, 256, 0, stream>>>(HBUF, gxw, gyw, GBUF);
      phy_bnstats_kernel<<<4, 256, 0, stream>>>(GBUF, STATS);
      phy_combine_kernel<<<eb, 256, 0, stream>>>(W0, t, HBUF, FBBUF, GBUF, KGBUF, STATS, bng, bnb, Wp);
    }
  }

  // ---- x = LN2(x + p1*dst2 + p2*phy) ----
  ln(W0, Wm, Wp, p1, p2, ln2g, ln2b, W0);

  // ---- FFN + LN3 -> d_out ----
  gemm(W0, w1, b1, nullptr, Wff, FF, DD, 1);
  gemm(Wff, w2, b2, nullptr, Wm, DD, FF, 0);
  ln(W0, Wm, nullptr, nullptr, nullptr, ln3g, ln3b, (float*)d_out);
}
